// MetricSelfAttention_71588514890324
// MI455X (gfx1250) — matrix operand order repaired and checked by reading, not compiled, not run on hardware
//
#include <hip/hip_runtime.h>
#include <hip/hip_bf16.h>

typedef __attribute__((ext_vector_type(16))) __bf16 v16bf;
typedef __attribute__((ext_vector_type(8)))  float  v8f;

// ---------------------------------------------------------------------------
// WMMA fragment helpers (CDNA5 wave32, V_WMMA_F32_16X16X32_BF16)
//  A 16x32 bf16 from row-major [m][k]: elements 0..7 <-> k=kh..kh+7,
//    8..15 <-> k=16+kh..16+kh+7, kh=(lane>>4)*8, m=lane&15.
//  B 32x16 bf16 from B^T layout [n][k] (k contiguous): lane&15=n,
//    same k order as A: elements 0..7 <-> k=kh..kh+7, 8..15 <-> k=16+kh..
//  C/D 16x16 f32: vgpr v, lane l -> row = v + (l>>4)*8, col = l&15.
// ---------------------------------------------------------------------------
__device__ inline v16bf frag_a(const __bf16* base, int ld) {
  int lane = threadIdx.x & 31;
  const __bf16* p = base + (lane & 15) * ld + ((lane >> 4) << 3);
  v16bf a;
  reinterpret_cast<uint4*>(&a)[0] = *reinterpret_cast<const uint4*>(p);
  reinterpret_cast<uint4*>(&a)[1] = *reinterpret_cast<const uint4*>(p + 16);
  return a;
}

__device__ inline v16bf frag_bt(const __bf16* base, int ld) {
  int lane = threadIdx.x & 31;
  const __bf16* p = base + (lane & 15) * ld + ((lane >> 4) << 3);
  v16bf b;
  reinterpret_cast<uint4*>(&b)[0] = *reinterpret_cast<const uint4*>(p);
  reinterpret_cast<uint4*>(&b)[1] = *reinterpret_cast<const uint4*>(p + 16);
  return b;
}

__device__ inline v8f wmma_bf16(v16bf a, v16bf b, v8f c) {
  return __builtin_amdgcn_wmma_f32_16x16x32_bf16(
      false, a, false, b, (short)0, c, false, false);
}

// ---------------------------------------------------------------------------
// Prep kernels (tiny): metric = P P^T (symmetric); Wp -> bf16;
// WoutT[c][n*64+k] = sum_j T_n[k,j] * Wm[c, n*64+j]  (transform+mixer fold)
// ---------------------------------------------------------------------------
__global__ void prep_metric(const float* __restrict__ P, __bf16* __restrict__ Mbf) {
  int n = blockIdx.x;
  const float* Pn = P + n * 64 * 64;
  for (int idx = threadIdx.x; idx < 4096; idx += 256) {
    int ii = idx >> 6, jj = idx & 63;
    float acc = 0.f;
    #pragma unroll 8
    for (int k = 0; k < 64; ++k) acc += Pn[ii * 64 + k] * Pn[jj * 64 + k];
    Mbf[n * 4096 + idx] = (__bf16)acc;
  }
}

__global__ void prep_cvt_bf16(const float* __restrict__ src, __bf16* __restrict__ dst, int nElem) {
  int i4 = (blockIdx.x * 256 + threadIdx.x) * 4;
  if (i4 + 3 < nElem) {
    float4 f = *reinterpret_cast<const float4*>(src + i4);
    dst[i4 + 0] = (__bf16)f.x; dst[i4 + 1] = (__bf16)f.y;
    dst[i4 + 2] = (__bf16)f.z; dst[i4 + 3] = (__bf16)f.w;
  }
}

__global__ void prep_wout(const float* __restrict__ T, const float* __restrict__ Wm,
                          __bf16* __restrict__ WoutT) {
  int idx = blockIdx.x * 256 + threadIdx.x;
  int c = idx >> 10, kk = idx & 1023;
  int n = kk >> 6, k = kk & 63;
  const float* Tn = T + (n * 64 + k) * 64;
  const float* Wr = Wm + (size_t)c * 1024 + n * 64;
  float acc = 0.f;
  #pragma unroll 8
  for (int j = 0; j < 64; ++j) acc += Tn[j] * Wr[j];
  WoutT[(size_t)c * 1024 + kk] = (__bf16)acc;
}

// ---------------------------------------------------------------------------
// 128x128 GEMM, BK=64, 8 waves, register->LDS double buffering,
// 1 barrier / 64-wide k-step (16 WMMA per wave between barriers).
// C[m][n] = sum_k A[m][k] * Bt[n][k]
// ---------------------------------------------------------------------------
template <bool A_F32, bool OUT_F32>
__global__ __launch_bounds__(256) void gemm128(const void* __restrict__ Ag,
                                               const __bf16* __restrict__ Btg,
                                               void* __restrict__ Cg,
                                               int M, int Ncol, int Kd) {
  __shared__ alignas(16) __bf16 sA[2][128 * 64];
  __shared__ alignas(16) __bf16 sB[2][128 * 64];
  int tid = threadIdx.x, lane = tid & 31, wid = tid >> 5;
  int m0 = blockIdx.y * 128, n0 = blockIdx.x * 128;
  int wm = (wid & 3) * 32, wn = (wid >> 2) * 64;

  const float*  Af = (const float*)Ag;
  const __bf16* Ab = (const __bf16*)Ag;

  float4 aRegF[8];  // A_F32: 32 floats per thread (128x64 tile / 256 thr)
  uint4  aRegB[4];  // !A_F32: 32 bf16 per thread
  uint4  bReg[4];   // 32 bf16 per thread

  auto loadTile = [&](int kb) {
    if constexpr (A_F32) {
      #pragma unroll
      for (int it = 0; it < 8; ++it) {
        int e = tid + it * 256;                  // 0..2047, float4 granules
        int r = e >> 4, cc = (e & 15) * 4;
        aRegF[it] = *reinterpret_cast<const float4*>(Af + (size_t)(m0 + r) * Kd + kb + cc);
      }
    } else {
      #pragma unroll
      for (int it = 0; it < 4; ++it) {
        int e = tid + it * 256;                  // 0..1023, 8-bf16 granules
        int r = e >> 3, cc = (e & 7) * 8;
        aRegB[it] = *reinterpret_cast<const uint4*>(Ab + (size_t)(m0 + r) * Kd + kb + cc);
      }
    }
    #pragma unroll
    for (int it = 0; it < 4; ++it) {
      int e = tid + it * 256;
      int r = e >> 3, cc = (e & 7) * 8;
      bReg[it] = *reinterpret_cast<const uint4*>(Btg + (size_t)(n0 + r) * Kd + kb + cc);
    }
  };

  auto storeTile = [&](int buf) {
    if constexpr (A_F32) {
      #pragma unroll
      for (int it = 0; it < 8; ++it) {
        int e = tid + it * 256;
        int r = e >> 4, cc = (e & 15) * 4;
        __bf16* d = sA[buf] + r * 64 + cc;
        d[0] = (__bf16)aRegF[it].x; d[1] = (__bf16)aRegF[it].y;
        d[2] = (__bf16)aRegF[it].z; d[3] = (__bf16)aRegF[it].w;
      }
    } else {
      #pragma unroll
      for (int it = 0; it < 4; ++it) {
        int e = tid + it * 256;
        int r = e >> 3, cc = (e & 7) * 8;
        *reinterpret_cast<uint4*>(sA[buf] + r * 64 + cc) = aRegB[it];
      }
    }
    #pragma unroll
    for (int it = 0; it < 4; ++it) {
      int e = tid + it * 256;
      int r = e >> 3, cc = (e & 7) * 8;
      *reinterpret_cast<uint4*>(sB[buf] + r * 64 + cc) = bReg[it];
    }
  };

  v8f c[2][4];
  #pragma unroll
  for (int t = 0; t < 2; ++t)
    #pragma unroll
    for (int u = 0; u < 4; ++u) c[t][u] = (v8f)(0.f);

  int nk = Kd >> 6;
  loadTile(0);
  storeTile(0);
  __syncthreads();

  for (int kt = 0; kt < nk; ++kt) {
    int cur = kt & 1;
    bool hasNext = kt + 1 < nk;
    if (hasNext) {
      loadTile((kt + 1) << 6);
      if (kt + 2 < nk) {  // pull k+2 lines toward GL2 (global_prefetch_b8)
        int r = tid & 127;
        if constexpr (A_F32)
          __builtin_prefetch(Af + (size_t)(m0 + r) * Kd + ((kt + 2) << 6), 0, 3);
        else
          __builtin_prefetch(Ab + (size_t)(m0 + r) * Kd + ((kt + 2) << 6), 0, 3);
        __builtin_prefetch(Btg + (size_t)(n0 + r) * Kd + ((kt + 2) << 6), 0, 3);
      }
    }

    #pragma unroll
    for (int ks = 0; ks < 2; ++ks) {
      v16bf a0 = frag_a(sA[cur] + (wm + 0) * 64 + ks * 32, 64);
      v16bf a1 = frag_a(sA[cur] + (wm + 16) * 64 + ks * 32, 64);
      #pragma unroll
      for (int u = 0; u < 4; ++u) {
        v16bf bf = frag_bt(sB[cur] + (wn + u * 16) * 64 + ks * 32, 64);
        c[0][u] = wmma_bf16(a0, bf, c[0][u]);
        c[1][u] = wmma_bf16(a1, bf, c[1][u]);
      }
    }

    if (hasNext) storeTile(cur ^ 1);
    __syncthreads();
  }

  int rl = (lane >> 4) * 8, cl = lane & 15;
  #pragma unroll
  for (int t = 0; t < 2; ++t)
    #pragma unroll
    for (int u = 0; u < 4; ++u)
      #pragma unroll
      for (int v = 0; v < 8; ++v) {
        int row = m0 + wm + t * 16 + rl + v;
        int col = n0 + wn + u * 16 + cl;
        if (OUT_F32)
          ((float*)Cg)[(size_t)row * Ncol + col] = c[t][u][v];
        else
          ((__bf16*)Cg)[(size_t)row * Ncol + col] = (__bf16)c[t][u][v];
      }
}

// ---------------------------------------------------------------------------
// Q' = P_head @ M_head  (metric fold), per (b, head, 128-row tile). bf16 out.
// ---------------------------------------------------------------------------
__global__ __launch_bounds__(256) void qprime_kernel(const __bf16* __restrict__ Pbf,
                                                     const __bf16* __restrict__ Mbf,
                                                     __bf16* __restrict__ Qbf) {
  __shared__ alignas(16) __bf16 sPi[128 * 64];
  __shared__ alignas(16) __bf16 sM[64 * 64];
  int tid = threadIdx.x, lane = tid & 31, wid = tid >> 5;
  int i = blockIdx.x, b = blockIdx.y >> 4, n = blockIdx.y & 15;
  const __bf16* Ph = Pbf + (size_t)b * 2048 * 1024 + n * 64;

  #pragma unroll
  for (int e = tid; e < 1024; e += 256) {
    int r = e >> 3, kc = (e & 7) * 8;
    *reinterpret_cast<uint4*>(sPi + r * 64 + kc) =
        *reinterpret_cast<const uint4*>(Ph + (size_t)(i * 128 + r) * 1024 + kc);
  }
  #pragma unroll
  for (int e = tid; e < 512; e += 256) {
    int r = e >> 3, kc = (e & 7) * 8;
    *reinterpret_cast<uint4*>(sM + r * 64 + kc) =
        *reinterpret_cast<const uint4*>(Mbf + n * 4096 + r * 64 + kc);
  }
  __syncthreads();

  v8f q[4];
  #pragma unroll
  for (int u = 0; u < 4; ++u) q[u] = (v8f)(0.f);
  #pragma unroll
  for (int ks = 0; ks < 2; ++ks) {
    v16bf a = frag_a(sPi + wid * 16 * 64 + ks * 32, 64);
    #pragma unroll
    for (int u = 0; u < 4; ++u)
      q[u] = wmma_bf16(a, frag_bt(sM + u * 16 * 64 + ks * 32, 64), q[u]);
  }

  int rl = (lane >> 4) * 8, cl = lane & 15;
  #pragma unroll
  for (int u = 0; u < 4; ++u)
    #pragma unroll
    for (int v = 0; v < 8; ++v)
      Qbf[((size_t)b * 2048 + i * 128 + wid * 16 + rl + v) * 1024 + n * 64 + u * 16 + cl] =
          (__bf16)q[u][v];
}

// ---------------------------------------------------------------------------
// Flash-style causal bilinear attention. Double-buffered P_j tiles,
// ONE barrier per column tile (S tile is wave-private by row ownership).
//   S = Q'_i @ P_j^T / 8, mask col<=row; acc += S @ P_j. Z written bf16.
// ---------------------------------------------------------------------------
__global__ __launch_bounds__(256) void flash_attn(const __bf16* __restrict__ Pbf,
                                                  const __bf16* __restrict__ Qbf,
                                                  __bf16* __restrict__ Zbf) {
  __shared__ alignas(16) __bf16 sQ[128 * 64];      // Q' tile (A layout)
  __shared__ alignas(16) __bf16 sS[128 * 64];      // S tile (wave-private rows)
  __shared__ alignas(16) __bf16 sPj[2][64 * 64];   // P_j [v][k]
  __shared__ alignas(16) __bf16 sPjT[2][64 * 64];  // P_j^T [k][v]

  int tid = threadIdx.x, lane = tid & 31, wid = tid >> 5;
  int i = blockIdx.x, b = blockIdx.y >> 4, n = blockIdx.y & 15;
  const __bf16* Ph = Pbf + (size_t)b * 2048 * 1024 + n * 64;
  const __bf16* Qh = Qbf + (size_t)b * 2048 * 1024 + n * 64;

  // load Q'_i tile and the jc=0 P tile (direct + transposed)
  #pragma unroll
  for (int e = tid; e < 1024; e += 256) {
    int r = e >> 3, kc = (e & 7) * 8;
    *reinterpret_cast<uint4*>(sQ + r * 64 + kc) =
        *reinterpret_cast<const uint4*>(Qh + (size_t)(i * 128 + r) * 1024 + kc);
  }
  #pragma unroll
  for (int it = 0; it < 2; ++it) {
    int e = tid + it * 256;
    int r = e >> 3, kc = (e & 7) * 8;
    union { uint4 u4; __bf16 h[8]; } t;
    t.u4 = *reinterpret_cast<const uint4*>(Ph + (size_t)r * 1024 + kc);
    *reinterpret_cast<uint4*>(sPj[0] + r * 64 + kc) = t.u4;
    #pragma unroll
    for (int x = 0; x < 8; ++x) sPjT[0][(kc + x) * 64 + r] = t.h[x];
  }
  __syncthreads();

  v8f acc[4];
  #pragma unroll
  for (int u = 0; u < 4; ++u) acc[u] = (v8f)(0.f);

  int rl = (lane >> 4) * 8, cl = lane & 15;
  int jmax = 2 * i + 1;

  for (int jc = 0; jc <= jmax; ++jc) {
    int cur = jc & 1;
    bool hasNext = jc < jmax;

    // register-prefetch next P_j tile
    uint4 pre[2];
    if (hasNext) {
      #pragma unroll
      for (int it = 0; it < 2; ++it) {
        int e = tid + it * 256;
        int r = e >> 3, kc = (e & 7) * 8;
        pre[it] = *reinterpret_cast<const uint4*>(
            Ph + (size_t)((jc + 1) * 64 + r) * 1024 + kc);
      }
    }

    // S = Q' @ P_j^T  (each wave: its 16 rows x 64 cols)
    v8f s[4];
    #pragma unroll
    for (int u = 0; u < 4; ++u) s[u] = (v8f)(0.f);
    #pragma unroll
    for (int ks = 0; ks < 2; ++ks) {
      v16bf a = frag_a(sQ + wid * 16 * 64 + ks * 32, 64);
      #pragma unroll
      for (int u = 0; u < 4; ++u)
        s[u] = wmma_bf16(a, frag_bt(sPj[cur] + u * 16 * 64 + ks * 32, 64), s[u]);
    }
    // mask (col<=row), scale 1/sqrt(64), requantize into wave-private S rows
    #pragma unroll
    for (int u = 0; u < 4; ++u)
      #pragma unroll
      for (int v = 0; v < 8; ++v) {
        int rloc = wid * 16 + rl + v, cloc = u * 16 + cl;
        float val = ((jc * 64 + cloc) <= (i * 128 + rloc)) ? s[u][v] * 0.125f : 0.0f;
        sS[rloc * 64 + cloc] = (__bf16)val;
      }

    // acc += S @ P_j  (contraction over this tile's 64 columns)
    #pragma unroll
    for (int ks = 0; ks < 2; ++ks) {
      v16bf a = frag_a(sS + wid * 16 * 64 + ks * 32, 64);
      #pragma unroll
      for (int u = 0; u < 4; ++u)
        acc[u] = wmma_bf16(a, frag_bt(sPjT[cur] + u * 16 * 64 + ks * 32, 64), acc[u]);
    }

    // commit next tile into the other buffer
    if (hasNext) {
      #pragma unroll
      for (int it = 0; it < 2; ++it) {
        int e = tid + it * 256;
        int r = e >> 3, kc = (e & 7) * 8;
        union { uint4 u4; __bf16 h[8]; } t;
        t.u4 = pre[it];
        *reinterpret_cast<uint4*>(sPj[cur ^ 1] + r * 64 + kc) = t.u4;
        #pragma unroll
        for (int x = 0; x < 8; ++x) sPjT[cur ^ 1][(kc + x) * 64 + r] = t.h[x];
      }
    }
    __syncthreads();
  }

  #pragma unroll
  for (int u = 0; u < 4; ++u)
    #pragma unroll
    for (int v = 0; v < 8; ++v)
      Zbf[((size_t)b * 2048 + i * 128 + wid * 16 + rl + v) * 1024 + n * 64 + u * 16 + cl] =
          (__bf16)acc[u][v];
}

// ---------------------------------------------------------------------------
extern "C" void kernel_launch(void* const* d_in, const int* in_sizes, int n_in,
                              void* d_out, int out_size, void* d_ws, size_t ws_size,
                              hipStream_t stream) {
  const float* X  = (const float*)d_in[0];   // [4,2048,1024]
  const float* Wp = (const float*)d_in[1];   // [1024,1024]
  const float* Pm = (const float*)d_in[2];   // [16,64,64]
  const float* Tr = (const float*)d_in[3];   // [16,64,64]
  const float* Wm = (const float*)d_in[4];   // [1024,1024]
  float* out = (float*)d_out;                // [4,2048,1024] f32

  char* ws = (char*)d_ws;
  size_t off = 0;
  __bf16* Pbf  = (__bf16*)(ws + off); off += (size_t)8192 * 1024 * 2;  // proj
  __bf16* Qbf  = (__bf16*)(ws + off); off += (size_t)8192 * 1024 * 2;  // proj @ metric
  __bf16* Zbf  = (__bf16*)(ws + off); off += (size_t)8192 * 1024 * 2;  // pre-mixer
  __bf16* Wpbf = (__bf16*)(ws + off); off += (size_t)1024 * 1024 * 2;
  __bf16* Mbf  = (__bf16*)(ws + off); off += (size_t)16 * 64 * 64 * 2;
  __bf16* Wo   = (__bf16*)(ws + off); off += (size_t)1024 * 1024 * 2;

  prep_metric<<<16, 256, 0, stream>>>(Pm, Mbf);
  prep_cvt_bf16<<<1024, 256, 0, stream>>>(Wp, Wpbf, 1024 * 1024);
  prep_wout<<<4096, 256, 0, stream>>>(Tr, Wm, Wo);

  // P = X @ Wp^T
  gemm128<true, false><<<dim3(8, 64), 256, 0, stream>>>(X, Wpbf, Pbf, 8192, 1024, 1024);

  // Q' = P @ M (per head)
  qprime_kernel<<<dim3(16, 64), 256, 0, stream>>>(Pbf, Mbf, Qbf);

  // causal bilinear attention
  flash_attn<<<dim3(16, 64), 256, 0, stream>>>(Pbf, Qbf, Zbf);

  // out = Z @ (T · Wm^T)
  gemm128<false, true><<<dim3(8, 64), 256, 0, stream>>>(Zbf, Wo, out, 8192, 1024, 1024);
}
